// SubjectLayers_51805895524871
// MI455X (gfx1250) — matrix operand order repaired and checked by reading, not compiled, not run on hardware
//
#include <hip/hip_runtime.h>

namespace {

constexpr int C_DIM = 270;   // channels (M and K)
constexpr int T_DIM = 1024;  // time (N)
constexpr int BM = 64;
constexpr int BN = 128;
constexpr int BK = 32;
constexpr int NKT = (C_DIM + BK - 1) / BK;  // 9 K-steps (last partial, zero-padded)
constexpr int ASTR = 40;  // LDS row stride (bf16 elems): 32 + 8 pad -> 80 B, 16B-aligned chunks
constexpr int BSTR = 40;

typedef __bf16 bf16;
typedef __bf16 v8bf  __attribute__((ext_vector_type(8)));
typedef __bf16 v16bf __attribute__((ext_vector_type(16)));
typedef float  v8f   __attribute__((ext_vector_type(8)));
typedef float  v4f   __attribute__((ext_vector_type(4)));

// Load a 16-element bf16 fragment from an LDS row as two 16-byte chunks
// at element offsets e0 / e1 (both 16B-aligned given ASTR/BSTR = 40).
__device__ __forceinline__ v16bf ld_frag(const bf16* rowp, int e0, int e1) {
    v8bf a = *reinterpret_cast<const v8bf*>(rowp + e0);
    v8bf b = *reinterpret_cast<const v8bf*>(rowp + e1);
    return __builtin_shufflevector(a, b, 0, 1, 2, 3, 4, 5, 6, 7,
                                         8, 9, 10, 11, 12, 13, 14, 15);
}

__device__ __forceinline__ v8f wmma_bf16(v16bf a, v16bf b, v8f c) {
    // (neg_a, A, neg_b, B, c_mod, C, reuse_a, reuse_b)
    return __builtin_amdgcn_wmma_f32_16x16x32_bf16(false, a, false, b,
                                                   (short)0, c, false, false);
}

} // namespace

__global__ __launch_bounds__(256)
void subject_layers_bf16x3(const float* __restrict__ x,
                           const int*   __restrict__ subjects,
                           const float* __restrict__ weights,
                           float* __restrict__ out)
{
    __shared__ bf16 sa_hi[BM * ASTR];
    __shared__ bf16 sa_lo[BM * ASTR];
    __shared__ bf16 sb_hi[BN * BSTR];
    __shared__ bf16 sb_lo[BN * BSTR];

    const int tid = threadIdx.x;
    const int b   = blockIdx.z;
    const int m0  = blockIdx.y * BM;   // output-channel tile (d)
    const int n0  = blockIdx.x * BN;   // time tile (t)

    const int subj = subjects[b];
    const float* __restrict__ wbase = weights + (size_t)subj * C_DIM * C_DIM;
    const float* __restrict__ xbase = x + (size_t)b * C_DIM * T_DIM;

    // ---- staging geometry (affine in tid) ----
    // A slab: 32(K) x 64(M) fp32 of w[k][m]; 8 scalars/thread, coalesced along m.
    const int am = tid & (BM - 1);          // 0..63
    const int ak = tid >> 6;                // 0..3  (+4 per unroll step)
    const bool am_ok = (m0 + am) < C_DIM;
    // B slab: 32(K) x 128(N) fp32 of x[k][n]; 4x float4/thread, coalesced along n.
    const int bn = (tid & 31) * 4;          // 0..124
    const int bk = tid >> 5;                // 0..7  (+8 per unroll step)

    float ra[8];
    v4f   rb[4];

    auto load_global = [&](int kt) {
        const int kbase = kt * BK;
#pragma unroll
        for (int i = 0; i < 8; ++i) {
            const int gk = kbase + ak + 4 * i;
            ra[i] = (am_ok && gk < C_DIM) ? wbase[gk * C_DIM + (m0 + am)] : 0.0f;
        }
#pragma unroll
        for (int i = 0; i < 4; ++i) {
            const int gk = kbase + bk + 8 * i;
            if (gk < C_DIM) {
                rb[i] = *reinterpret_cast<const v4f*>(&xbase[gk * T_DIM + (n0 + bn)]);
            } else {
                rb[i][0] = 0.f; rb[i][1] = 0.f; rb[i][2] = 0.f; rb[i][3] = 0.f;
            }
        }
    };

    auto stash_lds = [&]() {
#pragma unroll
        for (int i = 0; i < 8; ++i) {
            const int k = ak + 4 * i;
            const float v = ra[i];
            const bf16 h = (bf16)v;
            const bf16 l = (bf16)(v - (float)h);
            sa_hi[am * ASTR + k] = h;
            sa_lo[am * ASTR + k] = l;
        }
#pragma unroll
        for (int i = 0; i < 4; ++i) {
            const int k = bk + 8 * i;
#pragma unroll
            for (int j = 0; j < 4; ++j) {
                const float v = rb[i][j];
                const bf16 h = (bf16)v;
                const bf16 l = (bf16)(v - (float)h);
                sb_hi[(bn + j) * BSTR + k] = h;
                sb_lo[(bn + j) * BSTR + k] = l;
            }
        }
    };

    // ---- per-wave fragment geometry (wave32) ----
    const int lane = tid & 31;
    const int wv   = tid >> 5;        // wave id 0..7 -> N subtile
    const int lrow = lane & 15;
    const int half = lane >> 4;       // 0 or 1
    // A 16x32 bf16 layout: lane<16 -> K {0-7,16-23}; lane>=16 -> K {8-15,24-31}; M = lane%16
    const int ae0 = half * 8, ae1 = ae0 + 16;
    // B 32x16 bf16 layout: lane<16 -> K {0-7,16-23}; lane>=16 -> K {8-15,24-31}; N = lane%16
    const int be0 = half * 8, be1 = be0 + 16;
    const int brow = (wv * 16 + lrow) * BSTR;

    v8f acc[4];
#pragma unroll
    for (int mt = 0; mt < 4; ++mt)
#pragma unroll
        for (int v = 0; v < 8; ++v) acc[mt][v] = 0.0f;

    load_global(0);

    for (int kt = 0; kt < NKT; ++kt) {
        __syncthreads();                 // LDS readers of previous step done
        stash_lds();
        __syncthreads();
        if (kt + 1 < NKT) load_global(kt + 1);   // prefetch overlaps WMMAs below

        const v16bf bh = ld_frag(&sb_hi[brow], be0, be1);
        const v16bf bl = ld_frag(&sb_lo[brow], be0, be1);
#pragma unroll
        for (int mt = 0; mt < 4; ++mt) {
            const int arow = (mt * 16 + lrow) * ASTR;
            const v16bf ah = ld_frag(&sa_hi[arow], ae0, ae1);
            const v16bf al = ld_frag(&sa_lo[arow], ae0, ae1);
            acc[mt] = wmma_bf16(ah, bh, acc[mt]);   // hi*hi
            acc[mt] = wmma_bf16(al, bh, acc[mt]);   // lo*hi
            acc[mt] = wmma_bf16(ah, bl, acc[mt]);   // hi*lo
        }
    }

    // ---- store D tiles: VGPR v -> M = v + 8*half, N = lane%16 ----
    const int tcol = n0 + wv * 16 + lrow;
    const int mofs = half * 8;
#pragma unroll
    for (int mt = 0; mt < 4; ++mt) {
#pragma unroll
        for (int v = 0; v < 8; ++v) {
            const int d = m0 + mt * 16 + mofs + v;
            if (d < C_DIM) {
                __builtin_nontemporal_store(
                    acc[mt][v], &out[((size_t)b * C_DIM + d) * T_DIM + tcol]);
            }
        }
    }
}

extern "C" void kernel_launch(void* const* d_in, const int* in_sizes, int n_in,
                              void* d_out, int out_size, void* d_ws, size_t ws_size,
                              hipStream_t stream) {
    const float* x        = (const float*)d_in[0];
    const int*   subjects = (const int*)d_in[1];
    const float* weights  = (const float*)d_in[2];
    float*       out      = (float*)d_out;

    const int B = in_sizes[1];  // subjects has one entry per sample
    dim3 grid(T_DIM / BN, (C_DIM + BM - 1) / BM, B);
    subject_layers_bf16x3<<<grid, 256, 0, stream>>>(x, subjects, weights, out);
}
